// GIFNeuron_56410100466007
// MI455X (gfx1250) — matrix operand order repaired and checked by reading, not compiled, not run on hardware
//
#include <hip/hip_runtime.h>

// ---- CDNA5 WMMA types ----
typedef __attribute__((ext_vector_type(16))) __bf16 v16bf;
typedef __attribute__((ext_vector_type(8)))  __bf16 v8bf;
typedef __attribute__((ext_vector_type(4)))  __bf16 v4bf;
typedef __attribute__((ext_vector_type(8)))  float  v8f;

#define B_  32
#define S_  1024
#define D_  512
#define H_  1024
#define LPAD 520          // padded LDS row stride in bf16 elements (+8 => bank rotate 4/row)
#define NW  4             // waves per block
#define DECAY_F 0.9048374180359595f   // exp(-1/10)

__global__ __launch_bounds__(128) void gif_fused_kernel(
    const float* __restrict__ x,   // [B,S,D]
    const float* __restrict__ W,   // [H,D]
    const float* __restrict__ bia, // [H]
    float* __restrict__ out)       // spikes [B,S,H] ++ v [B,H] ++ theta [B,H]
{
    // double-buffered x slice (bf16), 2 x 16 rows x 512 (+pad)
    __shared__ alignas(16) __bf16 lA[2][16 * LPAD];

    const int tid  = threadIdx.x;
    const int wave = tid >> 5;
    const int lane = tid & 31;
    const int ln   = lane & 15;   // N column within tile / M row for A
    const int lh   = lane >> 4;   // half-select

    const int bt = blockIdx.x & 1;                             // b-tile (0/1)
    const int ht = (blockIdx.x >> 1) * (16 * NW) + wave * 16;  // this wave's H base

    // ---------- preload W fragments as B operand in this code's k order, same as A (lane: N=ln, K-range lh*16..+15) ----------
    v16bf wf[16];
    {
        const float* wrow = W + (size_t)(ht + ln) * D_;
        #pragma unroll
        for (int kc = 0; kc < 16; ++kc) {
            const float4* p = (const float4*)(wrow + kc * 32 + lh * 16);
            v16bf f;
            #pragma unroll
            for (int q = 0; q < 4; ++q) {
                float4 t = p[q];
                f[q*4+0] = (__bf16)t.x; f[q*4+1] = (__bf16)t.y;
                f[q*4+2] = (__bf16)t.z; f[q*4+3] = (__bf16)t.w;
            }
            wf[kc] = f;
        }
    }

    const float bn = bia[ht + ln];   // bias for this lane's neuron column

    // ---------- state in C-matrix layout: VGPR i, lane -> (M = i + 8*lh, N = ln) ----------
    v8f v, th;
    #pragma unroll
    for (int i = 0; i < 8; ++i) { v[i] = 0.0f; th[i] = 1.0f; }

    const int c4 = tid * 4;  // this thread's float4 column for staging (128 threads * 4 = 512)

    // ---------- staging registers: next timestep's x slice, already bf16-packed ----------
    v4bf stg[16];
    {   // prologue: load s = 0
        #pragma unroll
        for (int r = 0; r < 16; ++r) {
            const float4 xv = *(const float4*)(x + (((size_t)(bt*16 + r)) * S_ + 0) * D_ + c4);
            v4bf pk;
            pk[0] = (__bf16)xv.x; pk[1] = (__bf16)xv.y;
            pk[2] = (__bf16)xv.z; pk[3] = (__bf16)xv.w;
            stg[r] = pk;
        }
    }

    for (int s = 0; s < S_; ++s) {
        __bf16* buf = lA[s & 1];

        // ---- write staged slice to LDS (8B ds_store, conflict-free) ----
        #pragma unroll
        for (int r = 0; r < 16; ++r)
            *(v4bf*)(&buf[r * LPAD + c4]) = stg[r];
        __syncthreads();   // single barrier per step (safe via double buffer)

        // ---- issue next step's global loads early: overlap with WMMA chain ----
        if (s + 1 < S_) {
            #pragma unroll
            for (int r = 0; r < 16; ++r) {
                const float4 xv = *(const float4*)(x + (((size_t)(bt*16 + r)) * S_ + (s+1)) * D_ + c4);
                v4bf pk;
                pk[0] = (__bf16)xv.x; pk[1] = (__bf16)xv.y;
                pk[2] = (__bf16)xv.z; pk[3] = (__bf16)xv.w;
                stg[r] = pk;
            }
        }

        // ---- GEMM tile: C = sum_kc A(16x32) * B(32x16), 2-deep LDS pipeline ----
        const __bf16* ap = &buf[ln * LPAD + lh * 16];
        v8bf plo[2], phi[2];
        plo[0] = *(const v8bf*)(ap);        // chunk 0 lo  (ds_load_b128)
        phi[0] = *(const v8bf*)(ap + 8);    // chunk 0 hi
        v8f c = {};
        #pragma unroll
        for (int kc = 0; kc < 16; ++kc) {
            if (kc + 1 < 16) {              // issue chunk kc+1 before consuming chunk kc
                plo[(kc + 1) & 1] = *(const v8bf*)(ap + (kc + 1) * 32);
                phi[(kc + 1) & 1] = *(const v8bf*)(ap + (kc + 1) * 32 + 8);
            }
            v16bf a;
            #pragma unroll
            for (int e = 0; e < 8; ++e) {
                a[e]     = plo[kc & 1][e];
                a[e + 8] = phi[kc & 1][e];
            }
            c = __builtin_amdgcn_wmma_f32_16x16x32_bf16(
                    false, a, false, wf[kc], (short)0, c, false, false);
        }

        // ---- GIF neuron update + spike store ----
        #pragma unroll
        for (int i = 0; i < 8; ++i) {
            float vv  = v[i] * DECAY_F + (c[i] + bn);
            float t   = th[i];
            float lim = 16.0f * t;                         // L * theta * 2
            vv = fminf(fmaxf(vv, -lim), lim);
            float sp = floorf(fminf(fmaxf(vv / (t + 1e-6f), 0.0f), 8.0f));
            vv -= sp * t;
            t  += 0.01f * sp - 0.01f * (t - 1.0f);
            v[i]  = vv;
            th[i] = t;
            const int m = i + 8 * lh;                      // row within b-tile
            out[(((size_t)(bt*16 + m)) * S_ + s) * H_ + (ht + ln)] = sp;
        }
    }

    // ---- final state: v then theta, each [B,H], after spikes ----
    const size_t base = (size_t)B_ * S_ * H_;
    #pragma unroll
    for (int i = 0; i < 8; ++i) {
        const int m  = i + 8 * lh;
        const size_t bm = (size_t)(bt*16 + m);
        out[base +                     bm * H_ + (ht + ln)] = v[i];
        out[base + (size_t)B_ * H_ +   bm * H_ + (ht + ln)] = th[i];
    }
}

extern "C" void kernel_launch(void* const* d_in, const int* in_sizes, int n_in,
                              void* d_out, int out_size, void* d_ws, size_t ws_size,
                              hipStream_t stream) {
    (void)in_sizes; (void)n_in; (void)d_ws; (void)ws_size; (void)out_size;
    const float* x  = (const float*)d_in[0];   // [32,1024,512]
    const float* W  = (const float*)d_in[1];   // [1024,512]
    const float* b  = (const float*)d_in[2];   // [1024]
    float* out = (float*)d_out;

    dim3 grid(2 * (H_ / (16 * NW)));   // 2 b-tiles * 16 H-tiles = 32 blocks
    dim3 block(32 * NW);               // 4 waves (wave32)
    gif_fused_kernel<<<grid, block, 0, stream>>>(x, W, b, out);
}
